// RWA_38860864094758
// MI455X (gfx1250) — matrix operand order repaired and checked by reading, not compiled, not run on hardware
//
#include <hip/hip_runtime.h>

// RWA online-softmax RNN for MI455X (gfx1250, wave32, WMMA).
// Strategy:
//   1) Convert x + all weights to bf16 once (parallel).
//   2) Pre-pass GEMM (parallel, WMMA bf16): u_full = x@u_w^T + u_b,
//      gx = x@g_w[:,:F]^T + g_b, ax = x@a_w[:,:F]^T, shared A tiles.
//   3) 512 serial step kernels: gh/ah = h@{g,a}_w[:,F:]^T via WMMA
//      (weights L2-resident), fused fp32 online-softmax epilogue,
//      h written to bf16 history (double-buffered via history).
//   4) One big parallel WMMA GEMM: outs = h_hist @ o_w^T + o_b.
//   5) Tail kernel writes (n, d, h_f, a_new) output segments.

#define DEVFN __device__ __forceinline__

typedef __attribute__((ext_vector_type(16))) __bf16 v16bf;
typedef __attribute__((ext_vector_type(8)))  __bf16 v8bf;
typedef __attribute__((ext_vector_type(8)))  float  v8f;

constexpr int B_ = 64, T_ = 512, F_ = 256, C_ = 1024, K_ = 256;
constexpr int FC_ = F_ + C_;          // 1280
constexpr int WAVE = 32;

DEVFN __bf16 f2bf(float f) {
  union { float f; unsigned u; } in; in.f = f;
  unsigned r = in.u + 0x7FFFu + ((in.u >> 16) & 1u);   // round-to-nearest-even
  union { unsigned short s; __bf16 b; } out;
  out.s = (unsigned short)(r >> 16);
  return out.b;
}

// ---- WMMA helpers (contiguous-K 16-bit operand layouts) -------------------

// A tile (16xK=32, M rows striped over lanes; lanes<16: K 0..15;
// lanes>=16: K 16..31).
DEVFN v16bf loadA(const __bf16* __restrict__ base, size_t lda,
                  int m0, int k0, int lane) {
  int row = m0 + (lane & 15);
  int kb  = k0 + ((lane & 16) ? 16 : 0);
  const __bf16* p = base + (size_t)row * lda + kb;
  v8bf lo = *reinterpret_cast<const v8bf*>(p);        // K = kb .. kb+7
  v8bf hi = *reinterpret_cast<const v8bf*>(p + 8);    // K = kb+8 .. kb+15
  v16bf a;
#pragma unroll
  for (int i = 0; i < 8; ++i) { a[i] = lo[i]; a[8 + i] = hi[i]; }
  return a;
}

// B tile (K=32 x 16). Column n = weight row n (contiguous in K).
// lanes<16: column n0+lane, K k0..k0+15; lanes>=16: column n0+lane-16, K k0+16..k0+31.
DEVFN v16bf loadB(const __bf16* __restrict__ base, size_t ldb,
                  int n0, int k0, int lane) {
  int col = n0 + (lane & 15);
  int kb  = k0 + ((lane & 16) ? 16 : 0);
  return *reinterpret_cast<const v16bf*>(base + (size_t)col * ldb + kb);
}

DEVFN v8f wmma_bf16(v16bf a, v16bf b, v8f c) {
  return __builtin_amdgcn_wmma_f32_16x16x32_bf16(
      false, a, false, b, (short)0, c, false, false);
}

// ---- kernels ----------------------------------------------------------------

__global__ void cvt_bf16_kernel(const float* __restrict__ src,
                                __bf16* __restrict__ dst, size_t n) {
  size_t i = (size_t)blockIdx.x * blockDim.x + threadIdx.x;
  size_t stride = (size_t)gridDim.x * blockDim.x;
  for (; i < n; i += stride) dst[i] = f2bf(src[i]);
}

// h0 = h + tanh(s); copy state inputs into mutable fp32 state; emit s output.
__global__ void init_kernel(const float* __restrict__ s,
                            const float* __restrict__ n_in,
                            const float* __restrict__ d_in_,
                            const float* __restrict__ h_in,
                            const float* __restrict__ amax_in,
                            __bf16* __restrict__ h0_bf,
                            float* __restrict__ n_s,
                            float* __restrict__ d_s,
                            float* __restrict__ amax_s,
                            float* __restrict__ s_out) {
  int i = blockIdx.x * blockDim.x + threadIdx.x;     // over B*C
  if (i >= B_ * C_) return;
  int c = i & (C_ - 1);
  h0_bf[i]  = f2bf(h_in[i] + tanhf(s[c]));
  n_s[i]    = n_in[i];
  d_s[i]    = d_in_[i];
  amax_s[i] = amax_in[i];
  if (i < C_) s_out[i] = s[i];
}

// Pre-pass: for all (b,t): u_full = x@u_w^T+u_b ; gx = x@g_w[:, :F]^T+g_b ;
// ax = x@a_w[:, :F]^T.  M = B*T = 32768, N = C = 1024, K = F = 256.
__global__ void pre_gemm_kernel(const __bf16* __restrict__ x_bf,
                                const __bf16* __restrict__ u_wbf,
                                const __bf16* __restrict__ g_wbf,
                                const __bf16* __restrict__ a_wbf,
                                const float*  __restrict__ u_b,
                                const float*  __restrict__ g_b,
                                float* __restrict__ u_full,
                                float* __restrict__ gx,
                                float* __restrict__ ax) {
  int lane = threadIdx.x & (WAVE - 1);
  int tile = blockIdx.x * (blockDim.x / WAVE) + (threadIdx.x >> 5);
  int nt = tile & 63;        // 64 n-tiles
  int mt = tile >> 6;        // 2048 m-tiles
  if (mt >= (B_ * T_) / 16) return;
  int m0 = mt * 16, n0 = nt * 16;

  v8f cu = {}, cg = {}, ca = {};
#pragma unroll 2
  for (int k0 = 0; k0 < F_; k0 += 32) {
    v16bf a  = loadA(x_bf, F_, m0, k0, lane);
    v16bf bu = loadB(u_wbf, F_,  n0, k0, lane);
    v16bf bg = loadB(g_wbf, FC_, n0, k0, lane);   // x-part columns 0..F-1
    v16bf ba = loadB(a_wbf, FC_, n0, k0, lane);
    cu = wmma_bf16(a, bu, cu);
    cg = wmma_bf16(a, bg, cg);
    ca = wmma_bf16(a, ba, ca);
  }
  int col = n0 + (lane & 15);
  float ub = u_b[col], gb = g_b[col];
  int rowOff = (lane & 16) ? 8 : 0;
#pragma unroll
  for (int i = 0; i < 8; ++i) {
    size_t r = (size_t)(m0 + rowOff + i);        // r = b*T + t
    size_t p = r * C_ + col;
    u_full[p] = cu[i] + ub;
    gx[p]     = cg[i] + gb;
    ax[p]     = ca[i];
  }
}

// One serial timestep: gh/ah = h_prev @ {g,a}_w[:,F:]^T then fused
// online-softmax update; writes h_t (bf16) into t-major history.
__global__ void step_kernel(int t,
                            const __bf16* __restrict__ h0_bf,
                            __bf16* __restrict__ h_hist,      // [T][B][C]
                            const __bf16* __restrict__ g_wbf, // [C][FC]
                            const __bf16* __restrict__ a_wbf, // [C][FC]
                            const float* __restrict__ u_full, // [B*T][C]
                            const float* __restrict__ gx,
                            const float* __restrict__ ax,
                            float* __restrict__ n_s,
                            float* __restrict__ d_s,
                            float* __restrict__ amax_s) {
  int lane = threadIdx.x & (WAVE - 1);
  int tile = blockIdx.x * (blockDim.x / WAVE) + (threadIdx.x >> 5);
  int nt = tile & 63;           // 64 cell tiles
  int mt = tile >> 6;           // 4 batch tiles
  int m0 = mt * 16, n0 = nt * 16;

  const __bf16* hp = (t == 0) ? h0_bf
                              : h_hist + (size_t)(t - 1) * B_ * C_;
  int col = n0 + (lane & 15);
  const __bf16* bgRow = g_wbf + (size_t)col * FC_ + F_;  // h-part columns
  const __bf16* baRow = a_wbf + (size_t)col * FC_ + F_;
  int kOffB = (lane & 16) ? 16 : 0;

  v8f cg = {}, ca = {};
#pragma unroll 4
  for (int k0 = 0; k0 < C_; k0 += 32) {
    v16bf a  = loadA(hp, C_, m0, k0, lane);
    v16bf bg = *reinterpret_cast<const v16bf*>(bgRow + k0 + kOffB);
    v16bf ba = *reinterpret_cast<const v16bf*>(baRow + k0 + kOffB);
    if (k0 + 32 < C_) {   // stream next weight chunks (global_prefetch_b8)
      __builtin_prefetch(bgRow + k0 + 32 + kOffB, 0, 0);
      __builtin_prefetch(baRow + k0 + 32 + kOffB, 0, 0);
    }
    cg = wmma_bf16(a, bg, cg);
    ca = wmma_bf16(a, ba, ca);
  }

  int rowOff = (lane & 16) ? 8 : 0;
#pragma unroll
  for (int i = 0; i < 8; ++i) {
    int b = m0 + rowOff + i;
    size_t sidx = (size_t)b * C_ + col;                    // state index
    size_t pidx = ((size_t)b * T_ + t) * C_ + col;         // precomputed index
    float u = u_full[pidx];
    float g = gx[pidx] + cg[i];
    float a = ax[pidx] + ca[i];
    float z = u * tanhf(g);
    float am = amax_s[sidx];
    float an = fmaxf(am, a);
    float e1 = __expf(am - an);
    float e2 = __expf(a - an);
    float nn = n_s[sidx] * e1 + z * e2;
    float dd = d_s[sidx] * e1 + e2;
    n_s[sidx] = nn;
    d_s[sidx] = dd;
    amax_s[sidx] = an;
    h_hist[(size_t)t * B_ * C_ + sidx] = f2bf(tanhf(nn / dd));
  }
}

// outs = h_hist @ o_w^T + o_b.  M = T*B = 32768 (rows r = t*B + b),
// N = K_ = 256, Kdim = C = 1024.  Output reindexed to [b][t][k].
__global__ void out_gemm_kernel(const __bf16* __restrict__ h_hist,
                                const __bf16* __restrict__ o_wbf, // [K][C]
                                const float*  __restrict__ o_b,
                                float* __restrict__ outs) {
  int lane = threadIdx.x & (WAVE - 1);
  int tile = blockIdx.x * (blockDim.x / WAVE) + (threadIdx.x >> 5);
  int nt = tile & 15;            // 16 n-tiles
  int mt = tile >> 4;            // 2048 m-tiles
  if (mt >= (B_ * T_) / 16) return;
  int m0 = mt * 16, n0 = nt * 16;

  v8f c = {};
#pragma unroll 4
  for (int k0 = 0; k0 < C_; k0 += 32) {
    v16bf a = loadA(h_hist, C_, m0, k0, lane);
    v16bf b = loadB(o_wbf, C_, n0, k0, lane);
    c = wmma_bf16(a, b, c);
  }
  int kcol = n0 + (lane & 15);
  float bias = o_b[kcol];
  int rowOff = (lane & 16) ? 8 : 0;
#pragma unroll
  for (int i = 0; i < 8; ++i) {
    int r = m0 + rowOff + i;        // r = t*B + b
    int t = r >> 6;                 // B_ = 64
    int b = r & 63;
    outs[((size_t)b * T_ + t) * K_ + kcol] = c[i] + bias;
  }
}

// Final state outputs: n_t, d_t, h_f = tanh(n/d), a_new.
__global__ void final_state_kernel(const float* __restrict__ n_s,
                                   const float* __restrict__ d_s,
                                   const float* __restrict__ amax_s,
                                   float* __restrict__ n_out,
                                   float* __restrict__ d_out_,
                                   float* __restrict__ h_out,
                                   float* __restrict__ a_out) {
  int i = blockIdx.x * blockDim.x + threadIdx.x;
  if (i >= B_ * C_) return;
  float nn = n_s[i], dd = d_s[i];
  n_out[i] = nn;
  d_out_[i] = dd;
  h_out[i] = tanhf(nn / dd);
  a_out[i] = amax_s[i];
}

// ---- host launcher ----------------------------------------------------------

extern "C" void kernel_launch(void* const* d_in, const int* in_sizes, int n_in,
                              void* d_out, int out_size, void* d_ws, size_t ws_size,
                              hipStream_t stream) {
  (void)in_sizes; (void)n_in; (void)out_size; (void)ws_size;
  const float* x      = (const float*)d_in[0];
  const float* s      = (const float*)d_in[1];
  const float* n_in_  = (const float*)d_in[2];
  const float* d_in_  = (const float*)d_in[3];
  const float* h_in   = (const float*)d_in[4];
  const float* amax_i = (const float*)d_in[5];
  const float* u_w    = (const float*)d_in[6];
  const float* u_b    = (const float*)d_in[7];
  const float* g_w    = (const float*)d_in[8];
  const float* g_b    = (const float*)d_in[9];
  const float* a_w    = (const float*)d_in[10];
  const float* o_w    = (const float*)d_in[11];
  const float* o_b    = (const float*)d_in[12];

  // Output segments: (outs, s, n_t, d_t, h_f, a_new)
  float* outP = (float*)d_out;
  float* out_outs = outP;
  float* out_s    = out_outs + (size_t)B_ * T_ * K_;
  float* out_n    = out_s + C_;
  float* out_d    = out_n + (size_t)B_ * C_;
  float* out_h    = out_d + (size_t)B_ * C_;
  float* out_a    = out_h + (size_t)B_ * C_;

  // Workspace carve (256B aligned). Total ~490 MB.
  char* w = (char*)d_ws;
  size_t off = 0;
  auto carve = [&](size_t bytes) {
    void* p = w + off;
    off += (bytes + 255) & ~(size_t)255;
    return p;
  };
  __bf16* x_bf   = (__bf16*)carve((size_t)B_ * T_ * F_ * 2);
  __bf16* u_wbf  = (__bf16*)carve((size_t)C_ * F_ * 2);
  __bf16* g_wbf  = (__bf16*)carve((size_t)C_ * FC_ * 2);
  __bf16* a_wbf  = (__bf16*)carve((size_t)C_ * FC_ * 2);
  __bf16* o_wbf  = (__bf16*)carve((size_t)K_ * C_ * 2);
  __bf16* h0_bf  = (__bf16*)carve((size_t)B_ * C_ * 2);
  __bf16* h_hist = (__bf16*)carve((size_t)T_ * B_ * C_ * 2);
  float*  u_full = (float*)carve((size_t)B_ * T_ * C_ * 4);
  float*  gx     = (float*)carve((size_t)B_ * T_ * C_ * 4);
  float*  ax     = (float*)carve((size_t)B_ * T_ * C_ * 4);
  float*  n_s    = (float*)carve((size_t)B_ * C_ * 4);
  float*  d_s    = (float*)carve((size_t)B_ * C_ * 4);
  float*  amax_s = (float*)carve((size_t)B_ * C_ * 4);

  // 1) bf16 conversions
  cvt_bf16_kernel<<<2048, 256, 0, stream>>>(x,   x_bf,  (size_t)B_ * T_ * F_);
  cvt_bf16_kernel<<<256,  256, 0, stream>>>(u_w, u_wbf, (size_t)C_ * F_);
  cvt_bf16_kernel<<<512,  256, 0, stream>>>(g_w, g_wbf, (size_t)C_ * FC_);
  cvt_bf16_kernel<<<512,  256, 0, stream>>>(a_w, a_wbf, (size_t)C_ * FC_);
  cvt_bf16_kernel<<<256,  256, 0, stream>>>(o_w, o_wbf, (size_t)K_ * C_);

  // 2) init state + h0 + s output
  init_kernel<<<(B_ * C_ + 255) / 256, 256, 0, stream>>>(
      s, n_in_, d_in_, h_in, amax_i, h0_bf, n_s, d_s, amax_s, out_s);

  // 3) parallel x-projection pre-pass: 131072 tiles, 8 waves/block
  pre_gemm_kernel<<<16384, 256, 0, stream>>>(
      x_bf, u_wbf, g_wbf, a_wbf, u_b, g_b, u_full, gx, ax);

  // 4) serial recurrence: 256 tiles/step, 8 waves/block -> 32 blocks
  for (int t = 0; t < T_; ++t) {
    step_kernel<<<32, 256, 0, stream>>>(
        t, h0_bf, h_hist, g_wbf, a_wbf, u_full, gx, ax, n_s, d_s, amax_s);
  }

  // 5) final parallel output GEMM: 32768 tiles, 8 waves/block
  out_gemm_kernel<<<4096, 256, 0, stream>>>(h_hist, o_wbf, o_b, out_outs);

  // 6) state outputs
  final_state_kernel<<<(B_ * C_ + 255) / 256, 256, 0, stream>>>(
      n_s, d_s, amax_s, out_n, out_d, out_h, out_a);
}
